// MambaBlock_44409961841182
// MI455X (gfx1250) — matrix operand order repaired and checked by reading, not compiled, not run on hardware
//
#include <hip/hip_runtime.h>
#include <hip/hip_bf16.h>

typedef __attribute__((ext_vector_type(16))) _Float16 v16h;
typedef __attribute__((ext_vector_type(8)))  _Float16 v8h;
typedef __attribute__((ext_vector_type(2)))  _Float16 v2h;
typedef __attribute__((ext_vector_type(8)))  float    v8f;
typedef __attribute__((ext_vector_type(4)))  float    v4f;

#define D_MODEL 768
#define D_INNER 1536
#define D_STATE 16
#define D_CONV  4
#define DT_RANK 48
#define BATCH   4
#define SEQ     2048
#define NTOK    (BATCH*SEQ)             /* 8192 */
#define NC      16                      /* scan chunks */
#define LC      (SEQ/NC)                /* 128 per chunk */
#define XPROJ_N (DT_RANK + 2*D_STATE)   /* 80 */

__device__ __forceinline__ float sigmoidf_(float x){ return 1.0f/(1.0f+__expf(-x)); }
__device__ __forceinline__ float siluf_(float x){ return x*sigmoidf_(x); }
__device__ __forceinline__ float softplusf_(float x){ return (x>20.0f)?x:log1pf(__expf(x)); }

// ---------------------------------------------------------------------------
// fp32-in/fp32-out GEMM, fp16 WMMA multiply, fp32 accumulate.
// C[M,N] = A[M,K](lda) @ B[K,N](ldb).
// 256 threads = 8 waves; 128x128 C tile; each wave a 32x64 subtile = 2x4
// v_wmma_f32_16x16x32_f16 per 32-k chunk.  LDS is double-buffered (ping-pong)
// so the next chunk's global loads overlap the current chunk's WMMAs.
//
// LDS is written directly in WMMA fragment order (contiguous k order, same for A and B):
//   A (16x32 f16): lane = (m%16) + 16*(k/16),     elem j = k%16
//     -> 8 consecutive k (8-aligned) are 8 consecutive halfs: one b128 store.
//   B (32x16 f16): lane = (n%16) + 16*(k/16),     elem j = k%16
//     -> consecutive k are consecutive halfs: pack k-pairs, ds_store_b32.
// Fragment loads are aligned 32B LDS reads (2x ds_load_b128 per v16h).
//
// Fast-path (unchecked) requirements: M tiles fully (M=8192 always),
// N % 16 == 0, K % 8 == 0, lda/ldb % 4 == 0.  CHKK/CHKN add group guards.
// ---------------------------------------------------------------------------
template<bool CHKK, bool CHKN>
__global__ __launch_bounds__(256)
void gemm_wmma(const float* __restrict__ A, int lda,
               const float* __restrict__ B, int ldb,
               float* __restrict__ C, int ldc,
               int M, int N, int K)
{
    __shared__ __align__(32) _Float16 Ash[2][8][512];   // ping-pong, 8 row-blocks
    __shared__ __align__(32) _Float16 Bsh[2][8][512];   // ping-pong, 8 col-blocks

    const int tid  = threadIdx.x;
    const int lane = tid & 31;
    const int wv   = tid >> 5;      // 0..7
    const int wm   = wv >> 1;       // 0..3  (32-row strip)
    const int wn   = wv & 1;        // 0..1  (64-col strip)
    const int m0   = blockIdx.y * 128;
    const int n0   = blockIdx.x * 128;

    v8f acc[2][4] = {};

    // ---- tile staging (global fp32 -> fragment-order fp16 LDS) ------------
    auto stageA = [&](int kk, int buf) {
        #pragma unroll
        for (int g = tid; g < 512; g += 256) {          // 8 k per group
            int r  = g >> 2;
            int kq = (g & 3) * 8;
            int gr = m0 + r, gk = kk + kq;
            v8h hv;
            if (!CHKK || (gk + 8 <= K)) {
                v4f f0 = *(const v4f*)&A[(size_t)gr * lda + gk];
                v4f f1 = *(const v4f*)&A[(size_t)gr * lda + gk + 4];
                #pragma unroll
                for (int i = 0; i < 4; ++i) { hv[i] = (_Float16)f0[i]; hv[4+i] = (_Float16)f1[i]; }
            } else {
                #pragma unroll
                for (int i = 0; i < 8; ++i)
                    hv[i] = (_Float16)((gk + i < K) ? A[(size_t)gr * lda + gk + i] : 0.0f);
            }
            int half = kq >> 4;
            int jb   = kq & 15;
            *(v8h*)&Ash[buf][r >> 4][((r & 15) + 16*half)*16 + jb] = hv;
        }
    };
    auto stageB = [&](int kk, int buf) {
        #pragma unroll
        for (int g = tid; g < 512; g += 256) {          // 2 k x 4 n per group
            int k2 = (g >> 5) << 1;                     // even k
            int n4 = (g & 31) * 4;
            int gk = kk + k2, gn0 = n0 + n4;
            float f0[4], f1[4];
            bool kin0 = !CHKK || (gk     < K);
            bool kin1 = !CHKK || (gk + 1 < K);
            bool nin  = !CHKN || (gn0 + 4 <= N);
            if (kin0 && nin) {
                v4f fv = *(const v4f*)&B[(size_t)gk * ldb + gn0];
                #pragma unroll
                for (int i = 0; i < 4; ++i) f0[i] = fv[i];
            } else {
                #pragma unroll
                for (int i = 0; i < 4; ++i)
                    f0[i] = (kin0 && (!CHKN || (gn0 + i < N))) ? B[(size_t)gk * ldb + gn0 + i] : 0.0f;
            }
            if (kin1 && nin) {
                v4f fv = *(const v4f*)&B[(size_t)(gk + 1) * ldb + gn0];
                #pragma unroll
                for (int i = 0; i < 4; ++i) f1[i] = fv[i];
            } else {
                #pragma unroll
                for (int i = 0; i < 4; ++i)
                    f1[i] = (kin1 && (!CHKN || (gn0 + i < N))) ? B[(size_t)(gk + 1) * ldb + gn0 + i] : 0.0f;
            }
            int half = k2 >> 4;
            int j    = k2 & 15;                          // even -> 4B aligned pair
            #pragma unroll
            for (int i = 0; i < 4; ++i) {
                int n = n4 + i;
                v2h pr = { (_Float16)f0[i], (_Float16)f1[i] };
                *(v2h*)&Bsh[buf][n >> 4][((n & 15) + 16*half)*16 + j] = pr;
            }
        }
    };

    const int nk = (K + 31) / 32;
    stageA(0, 0);
    stageB(0, 0);
    __syncthreads();

    for (int c = 0; c < nk; ++c) {
        const int buf = c & 1;
        // software pipeline: stage chunk c+1 into the other buffer while the
        // WMMAs below chew on chunk c (loads issue before any wmma waits).
        if (c + 1 < nk) {
            stageA((c + 1) * 32, buf ^ 1);
            stageB((c + 1) * 32, buf ^ 1);
        }
        // speculative prefetch two chunks ahead (global_prefetch_b8)
        if (c + 2 < nk)
            __builtin_prefetch(&A[(size_t)(m0 + (tid >> 1)) * lda + (c + 2) * 32], 0, 0);

        v16h a0 = *(const v16h*)&Ash[buf][wm*2 + 0][lane*16];
        v16h a1 = *(const v16h*)&Ash[buf][wm*2 + 1][lane*16];
        #pragma unroll
        for (int jf = 0; jf < 4; ++jf) {
            v16h bf = *(const v16h*)&Bsh[buf][wn*4 + jf][lane*16];
            acc[0][jf] = __builtin_amdgcn_wmma_f32_16x16x32_f16(false, a0, false, bf, (short)0, acc[0][jf], false, false);
            acc[1][jf] = __builtin_amdgcn_wmma_f32_16x16x32_f16(false, a1, false, bf, (short)0, acc[1][jf], false, false);
        }
        __syncthreads();   // staging of buf^1 done; compute reads of buf done
    }

    // C/D layout: VGPR v -> M = v + 8*(lane/16), N = lane%16.
    // N is always a multiple of 16 -> guard at fragment granularity only.
    // Lanes 0..15 cover 16 consecutive cols of one row -> coalesced segments.
    const int rbase = 8 * (lane >> 4);
    const int cn    = lane & 15;
    #pragma unroll
    for (int i = 0; i < 2; ++i)
        #pragma unroll
        for (int jf = 0; jf < 4; ++jf) {
            int colb = n0 + (wn*4 + jf)*16;
            if (CHKN && colb >= N) continue;
            int col = colb + cn;
            size_t rowb = (size_t)m0 + (wm*2 + i)*16 + rbase;
            #pragma unroll
            for (int v = 0; v < 8; ++v)
                C[(rowb + v) * ldc + col] = acc[i][jf][v];
        }
}

// ---------------------------------------------------------------------------
// Depthwise causal conv (width 4) + bias + SiLU, 4 channels per thread
// (float4 reads of the dominant xr stream).  x_in = xr[:, 0:1536].
// ---------------------------------------------------------------------------
__global__ __launch_bounds__(256)
void conv_silu(const float* __restrict__ xr, const float* __restrict__ cw,
               const float* __restrict__ cb, float* __restrict__ xc)
{
    size_t gid = (size_t)blockIdx.x * 256 + threadIdx.x;       // one per 4 channels
    if (gid >= (size_t)NTOK * (D_INNER/4)) return;
    int d4   = (int)(gid % (D_INNER/4)) * 4;
    int tlin = (int)(gid / (D_INNER/4));
    int b = tlin / SEQ, t = tlin % SEQ;

    float acc[4];
    #pragma unroll
    for (int c = 0; c < 4; ++c) acc[c] = cb[d4 + c];

    #pragma unroll
    for (int i = 0; i < D_CONV; ++i) {
        int tt = t - (D_CONV - 1) + i;
        if (tt >= 0) {
            v4f v = *(const v4f*)&xr[((size_t)(b*SEQ + tt)) * (2*D_INNER) + d4];
            #pragma unroll
            for (int c = 0; c < 4; ++c) acc[c] += cw[(d4 + c)*D_CONV + i] * v[c];
        }
    }
    v4f o;
    #pragma unroll
    for (int c = 0; c < 4; ++c) o[c] = siluf_(acc[c]);
    *(v4f*)&xc[(size_t)tlin * D_INNER + d4] = o;
}

// ---------------------------------------------------------------------------
// Chunked selective scan, pass 1: per-chunk local state S and decay product P
// with h0 = 0.  Grid: (NC, D_INNER/256, BATCH), block 256 (one channel/thread).
// B_t/C_t (32 floats/step) staged in LDS, shared by all 256 channels.
// ---------------------------------------------------------------------------
__global__ __launch_bounds__(256)
void scan_part1(const float* __restrict__ dtraw, const float* __restrict__ xdbl,
                const float* __restrict__ xc, const float* __restrict__ b_dt,
                const float* __restrict__ log_A,
                float* __restrict__ Sst, float* __restrict__ Pst)
{
    __shared__ float sBC[LC][32];
    const int d  = blockIdx.y * 256 + threadIdx.x;
    const int b  = blockIdx.z;
    const int j  = blockIdx.x;
    const int t0 = j * LC;

    for (int idx = threadIdx.x; idx < LC*32; idx += 256) {
        int tl = idx >> 5, c = idx & 31;
        sBC[tl][c] = xdbl[((size_t)(b*SEQ + t0 + tl)) * XPROJ_N + DT_RANK + c];
    }
    __syncthreads();

    float A[D_STATE], h[D_STATE], p[D_STATE];
    #pragma unroll
    for (int n = 0; n < D_STATE; ++n) {
        A[n] = -__expf(log_A[d*D_STATE + n]);
        h[n] = 0.0f; p[n] = 1.0f;
    }
    const float bdt = b_dt[d];

    for (int tl = 0; tl < LC; ++tl) {
        size_t rt = (size_t)(b*SEQ + t0 + tl);
        float dt = softplusf_(dtraw[rt*D_INNER + d] + bdt);
        float x  = xc[rt*D_INNER + d];
        float dx = dt * x;
        #pragma unroll
        for (int n = 0; n < D_STATE; ++n) {
            float a = __expf(A[n]*dt);
            h[n] = a*h[n] + dx*sBC[tl][n];
            p[n] *= a;
        }
    }
    size_t base = (((size_t)b*NC + j)*D_INNER + d)*D_STATE;
    #pragma unroll
    for (int n = 0; n < D_STATE; ++n) { Sst[base+n] = h[n]; Pst[base+n] = p[n]; }
}

// ---------------------------------------------------------------------------
// Carry combine: sequential over NC=16 chunk summaries, parallel over (b,d,n).
// ---------------------------------------------------------------------------
__global__ __launch_bounds__(256)
void scan_carry(const float* __restrict__ S, const float* __restrict__ P,
                float* __restrict__ H)
{
    int idx = blockIdx.x*256 + threadIdx.x;
    if (idx >= BATCH*D_INNER*D_STATE) return;
    int b  = idx / (D_INNER*D_STATE);
    int dn = idx % (D_INNER*D_STATE);
    float h = 0.0f;
    for (int j = 0; j < NC; ++j) {
        size_t off = (((size_t)b*NC + j) * (D_INNER*D_STATE)) + dn;
        H[off] = h;
        h = P[off]*h + S[off];
    }
}

// ---------------------------------------------------------------------------
// Pass 3: replay each chunk from corrected initial state, emit
// yg = (y + D*xc) * silu(res) fused epilogue.
// ---------------------------------------------------------------------------
__global__ __launch_bounds__(256)
void scan_part2(const float* __restrict__ dtraw, const float* __restrict__ xdbl,
                const float* __restrict__ xc, const float* __restrict__ xr,
                const float* __restrict__ b_dt, const float* __restrict__ log_A,
                const float* __restrict__ Dvec, const float* __restrict__ H,
                float* __restrict__ yg)
{
    __shared__ float sBC[LC][32];
    const int d  = blockIdx.y * 256 + threadIdx.x;
    const int b  = blockIdx.z;
    const int j  = blockIdx.x;
    const int t0 = j * LC;

    for (int idx = threadIdx.x; idx < LC*32; idx += 256) {
        int tl = idx >> 5, c = idx & 31;
        sBC[tl][c] = xdbl[((size_t)(b*SEQ + t0 + tl)) * XPROJ_N + DT_RANK + c];
    }
    __syncthreads();

    float A[D_STATE], h[D_STATE];
    size_t hbase = (((size_t)b*NC + j)*D_INNER + d)*D_STATE;
    #pragma unroll
    for (int n = 0; n < D_STATE; ++n) {
        A[n] = -__expf(log_A[d*D_STATE + n]);
        h[n] = H[hbase + n];
    }
    const float bdt = b_dt[d];
    const float Dd  = Dvec[d];

    for (int tl = 0; tl < LC; ++tl) {
        size_t rt = (size_t)(b*SEQ + t0 + tl);
        float dt = softplusf_(dtraw[rt*D_INNER + d] + bdt);
        float x  = xc[rt*D_INNER + d];
        float dx = dt * x;
        float y  = 0.0f;
        #pragma unroll
        for (int n = 0; n < D_STATE; ++n) {
            float a = __expf(A[n]*dt);
            h[n] = a*h[n] + dx*sBC[tl][n];
            y   += h[n] * sBC[tl][16 + n];
        }
        float res = xr[rt*(2*D_INNER) + D_INNER + d];
        yg[rt*D_INNER + d] = (y + Dd*x) * siluf_(res);
    }
}

// ---------------------------------------------------------------------------
extern "C" void kernel_launch(void* const* d_in, const int* in_sizes, int n_in,
                              void* d_out, int out_size, void* d_ws, size_t ws_size,
                              hipStream_t stream)
{
    const float* x       = (const float*)d_in[0];
    const float* W_in    = (const float*)d_in[1];
    const float* conv_w  = (const float*)d_in[2];
    const float* conv_b  = (const float*)d_in[3];
    const float* W_xproj = (const float*)d_in[4];
    const float* W_dt    = (const float*)d_in[5];
    const float* b_dt    = (const float*)d_in[6];
    const float* log_A   = (const float*)d_in[7];
    const float* Dv      = (const float*)d_in[8];
    const float* W_out   = (const float*)d_in[9];
    float* out = (float*)d_out;

    float* ws = (float*)d_ws;
    size_t off = 0;
    float* xr    = ws + off; off += (size_t)NTOK * 2 * D_INNER;
    float* xc    = ws + off; off += (size_t)NTOK * D_INNER;
    float* xdbl  = ws + off; off += (size_t)NTOK * XPROJ_N;
    float* dtraw = ws + off; off += (size_t)NTOK * D_INNER;
    float* yg    = ws + off; off += (size_t)NTOK * D_INNER;
    float* Sst   = ws + off; off += (size_t)BATCH*NC*D_INNER*D_STATE;
    float* Pst   = ws + off; off += (size_t)BATCH*NC*D_INNER*D_STATE;
    float* Hst   = ws + off; off += (size_t)BATCH*NC*D_INNER*D_STATE;
    (void)ws_size; (void)in_sizes; (void)n_in; (void)out_size;

    dim3 gblk(256);

    // 1) xr = x @ W_in   [8192x768]·[768x3072]  (fully aligned)
    gemm_wmma<false,false><<<dim3((2*D_INNER + 127)/128, (NTOK + 127)/128), gblk, 0, stream>>>(
        x, D_MODEL, W_in, 2*D_INNER, xr, 2*D_INNER, NTOK, 2*D_INNER, D_MODEL);

    // 2) depthwise conv + SiLU
    conv_silu<<<(unsigned)(((size_t)NTOK*(D_INNER/4) + 255)/256), 256, 0, stream>>>(
        xr, conv_w, conv_b, xc);

    // 3) x_dbl = xc @ W_xproj   [8192x1536]·[1536x80]  (N=80 checked)
    gemm_wmma<false,true><<<dim3((XPROJ_N + 127)/128, (NTOK + 127)/128), gblk, 0, stream>>>(
        xc, D_INNER, W_xproj, XPROJ_N, xdbl, XPROJ_N, NTOK, XPROJ_N, D_INNER);

    // 4) dtraw = x_dbl[:, :48] @ W_dt   (lda=80, K=48 checked)
    gemm_wmma<true,false><<<dim3((D_INNER + 127)/128, (NTOK + 127)/128), gblk, 0, stream>>>(
        xdbl, XPROJ_N, W_dt, D_INNER, dtraw, D_INNER, NTOK, D_INNER, DT_RANK);

    // 5) chunked selective scan (2-pass + carry) with fused epilogue
    dim3 sg(NC, D_INNER/256, BATCH);
    scan_part1<<<sg, 256, 0, stream>>>(dtraw, xdbl, xc, b_dt, log_A, Sst, Pst);
    scan_carry<<<(BATCH*D_INNER*D_STATE + 255)/256, 256, 0, stream>>>(Sst, Pst, Hst);
    scan_part2<<<sg, 256, 0, stream>>>(dtraw, xdbl, xc, xr, b_dt, log_A, Dv, Hst, yg);

    // 6) out = yg @ W_out   [8192x1536]·[1536x768]  (fully aligned)
    gemm_wmma<false,false><<<dim3((D_MODEL + 127)/128, (NTOK + 127)/128), gblk, 0, stream>>>(
        yg, D_INNER, W_out, D_MODEL, out, D_MODEL, NTOK, D_MODEL, D_INNER);
}
